// LeanSelfAttention_77670188581465
// MI455X (gfx1250) — matrix operand order repaired and checked by reading, not compiled, not run on hardware
//
#include <hip/hip_runtime.h>

#define HID    1024
#define NHEADS 16
#define HDIM   64
#define SEQ    2048
#define BATCH  4
#define BS     (BATCH * SEQ)   // 8192 rows

typedef __attribute__((ext_vector_type(16))) __bf16 v16bf;
typedef __attribute__((ext_vector_type(8)))  __bf16 v8bf;
typedef __attribute__((ext_vector_type(4)))  __bf16 v4bf;
typedef __attribute__((ext_vector_type(8)))  float  v8f;
typedef __attribute__((ext_vector_type(4)))  float  v4f;

// ---------- helpers ----------

static __device__ __forceinline__ __bf16 to_bf16(float f) {
  unsigned u = __builtin_bit_cast(unsigned, f);
  u += 0x7FFFu + ((u >> 16) & 1u);              // round-to-nearest-even
  unsigned short h = (unsigned short)(u >> 16);
  return __builtin_bit_cast(__bf16, h);
}

// A fragment 16x32 bf16: lane m = lane&15, hi = lane>>4; element h holds
// K = (h>>3)*16 + hi*8 + (h&7)  -> two contiguous 16B loads per lane.
static __device__ __forceinline__ v16bf load_frag_a(const __bf16* p, int ld) {
  int lane = threadIdx.x & 31;
  int m = lane & 15, hi = lane >> 4;
  const __bf16* q = p + (size_t)m * ld + hi * 8;
  v8bf lo = *(const v8bf*)(q);
  v8bf hv = *(const v8bf*)(q + 16);
  v16bf r;
#pragma unroll
  for (int i = 0; i < 8; ++i) { r[i] = lo[i]; r[i + 8] = hv[i]; }
  return r;
}

// B fragment 32x16 bf16 from K-contiguous (transposed) storage:
// lane n = lane&15, hi = lane>>4; element h holds
// K = (h>>3)*16 + hi*8 + (h&7)  -> two contiguous 16B loads per lane.
static __device__ __forceinline__ v16bf load_frag_b(const __bf16* p, int ld) {
  int lane = threadIdx.x & 31;
  int n = lane & 15, hi = lane >> 4;
  const __bf16* q = p + (size_t)n * ld + hi * 8;
  v8bf lo = *(const v8bf*)(q);
  v8bf hv = *(const v8bf*)(q + 16);
  v16bf r;
#pragma unroll
  for (int i = 0; i < 8; ++i) { r[i] = lo[i]; r[i + 8] = hv[i]; }
  return r;
}

// per-lane base address of a B fragment (for prefetch)
static __device__ __forceinline__ const __bf16* frag_b_ptr(const __bf16* p, int ld) {
  int lane = threadIdx.x & 31;
  return p + (size_t)(lane & 15) * ld + (lane >> 4) * 16;
}

static __device__ __forceinline__ v8f wmma_bf16(v16bf a, v16bf b, v8f c) {
  return __builtin_amdgcn_wmma_f32_16x16x32_bf16(false, a, false, b,
                                                 (short)0, c, false, false);
}

// one k-step's worth of fragments for a 16x64 tile
struct Frags { v16bf a; v16bf b[4]; };

static __device__ __forceinline__ Frags load_step(const __bf16* arow,
                                                  const __bf16* brow, int k0) {
  Frags f;
  f.a = load_frag_a(arow + k0, HID);
#pragma unroll
  for (int t = 0; t < 4; ++t)
    f.b[t] = load_frag_b(brow + (size_t)t * 16 * HID + k0, HID);
  return f;
}

static __device__ __forceinline__ void step_wmma(v8f acc[4], const Frags& f) {
#pragma unroll
  for (int t = 0; t < 4; ++t) acc[t] = wmma_bf16(f.a, f.b[t], acc[t]);
}

// ---------- kernel 1: f32 weight -> transposed bf16 ----------

__global__ void cvt_transpose(const float* __restrict__ w, __bf16* __restrict__ wt,
                              int K, int N) {
  int idx = blockIdx.x * 256 + threadIdx.x;
  if (idx >= K * N) return;
  int k = idx / N, n = idx - k * N;
  wt[(size_t)n * K + k] = to_bf16(w[idx]);
}

// ---------- kernel 2: LayerNorm -> bf16 (one wave per row) ----------

__global__ void __launch_bounds__(256) ln_kernel(
    const float* __restrict__ x, const float* __restrict__ g,
    const float* __restrict__ be, __bf16* __restrict__ o) {
  int lane = threadIdx.x & 31;
  int wid  = threadIdx.x >> 5;
  int row  = blockIdx.x * 8 + wid;
  const float* xr = x + (size_t)row * HID;
  v4f vals[8];
  float s = 0.f, s2 = 0.f;
#pragma unroll
  for (int i = 0; i < 8; ++i) {
    v4f v = *(const v4f*)(xr + (i * 32 + lane) * 4);
    vals[i] = v;
#pragma unroll
    for (int c = 0; c < 4; ++c) { s += v[c]; s2 += v[c] * v[c]; }
  }
#pragma unroll
  for (int m = 16; m >= 1; m >>= 1) {
    s  += __shfl_xor(s,  m, 32);
    s2 += __shfl_xor(s2, m, 32);
  }
  float mean = s * (1.f / HID);
  float var  = s2 * (1.f / HID) - mean * mean;
  float rstd = rsqrtf(var + 1e-12f);
  __bf16* orow = o + (size_t)row * HID;
#pragma unroll
  for (int i = 0; i < 8; ++i) {
    int col = (i * 32 + lane) * 4;
    v4f gv = *(const v4f*)(g + col);
    v4f bv = *(const v4f*)(be + col);
    v4bf ov;
#pragma unroll
    for (int c = 0; c < 4; ++c)
      ov[c] = to_bf16((vals[i][c] - mean) * rstd * gv[c] + bv[c]);
    *(v4bf*)(orow + col) = ov;
  }
}

// ---------- kernel 3: QKV GEMM (16x64 tile per wave, ping-pong pipeline) ----------
// writes Q,K -> [B,H,S,D] bf16 ; V -> transposed [B,H,D,S] bf16

__global__ void __launch_bounds__(256, 1) qkv_gemm(
    const __bf16* __restrict__ xln, const __bf16* __restrict__ wt,
    const float* __restrict__ bias,
    __bf16* __restrict__ qo, __bf16* __restrict__ ko,
    __bf16* __restrict__ vto) {
  int wave = (blockIdx.x * blockDim.x + threadIdx.x) >> 5;
  int mt = wave / 48;                 // 512 M-tiles
  int ns = wave - mt * 48;            // 48 N-strips of 64
  int lane = threadIdx.x & 31;
  int n_l = lane & 15, hi = lane >> 4;

  v8f acc[4] = {};
  const __bf16* arow = xln + (size_t)mt * 16 * HID;
  const __bf16* brow = wt + (size_t)ns * 64 * HID;

  // ping-pong double buffer: loads write directly into the loop-carried
  // buffer registers, no inter-buffer copies.
  Frags f0 = load_step(arow, brow, 0);
#pragma unroll 1
  for (int k0 = 32; k0 < HID - 32; k0 += 64) {
    Frags f1 = load_step(arow, brow, k0);
    step_wmma(acc, f0);
    f0 = load_step(arow, brow, k0 + 32);
    step_wmma(acc, f1);
  }
  {
    Frags f1 = load_step(arow, brow, HID - 32);
    step_wmma(acc, f0);
    step_wmma(acc, f1);
  }

#pragma unroll
  for (int t = 0; t < 4; ++t) {
    int n = ns * 64 + t * 16 + n_l;
    float bz = bias[n];
    int chunk = n >> 10;              // 0=Q 1=K 2=V
    int nn = n & 1023;
    int head = nn >> 6, d = nn & 63;
#pragma unroll
    for (int r = 0; r < 8; ++r) {
      int row = mt * 16 + hi * 8 + r;
      int b_ = row >> 11;             // / SEQ
      int s_ = row & (SEQ - 1);
      __bf16 v = to_bf16(acc[t][r] + bz);
      size_t qk = (((size_t)b_ * NHEADS + head) * SEQ + s_) * HDIM + d;
      if (chunk == 0)      qo[qk] = v;
      else if (chunk == 1) ko[qk] = v;
      else vto[(((size_t)b_ * NHEADS + head) * HDIM + d) * SEQ + s_] = v;
    }
  }
}

// ---------- kernel 4: flash attention (wave owns 16 q-rows) ----------

__global__ void __launch_bounds__(128, 1) attn_kernel(
    const __bf16* __restrict__ q, const __bf16* __restrict__ k,
    const __bf16* __restrict__ vt, const float* __restrict__ mask,
    __bf16* __restrict__ out) {
  __shared__ __align__(16) __bf16 Pb[4][16 * 32];
  int wid  = threadIdx.x >> 5;
  int lane = threadIdx.x & 31;
  int n_l = lane & 15, hi = lane >> 4;

  int blk = blockIdx.x;
  int qb  = blk & 31;                 // SEQ/64 q-blocks
  int bh  = blk >> 5;                 // b*NHEADS + h
  int b_  = bh >> 4;
  int qbase = qb * 64 + wid * 16;

  const __bf16* qp = q  + ((size_t)bh * SEQ + qbase) * HDIM;
  const __bf16* kp = k  + (size_t)bh * SEQ * HDIM;
  const __bf16* vp = vt + (size_t)bh * HDIM * SEQ;
  const float*  mp = mask + (size_t)b_ * SEQ;

  v16bf a0 = load_frag_a(qp, HDIM);        // K-dim 0..31 of d
  v16bf a1 = load_frag_a(qp + 32, HDIM);   // K-dim 32..63

  float rmax[8], rsum[8];
#pragma unroll
  for (int r = 0; r < 8; ++r) { rmax[r] = -3.0e38f; rsum[r] = 0.f; }
  v8f o[4] = {};
  __bf16* P = &Pb[wid][0];

#pragma unroll 1
  for (int j = 0; j < SEQ; j += 32) {
    v8f s0 = {}, s1 = {};
    s0 = wmma_bf16(a0, load_frag_b(kp + (size_t)j * HDIM, HDIM), s0);
    s0 = wmma_bf16(a1, load_frag_b(kp + (size_t)j * HDIM + 32, HDIM), s0);
    s1 = wmma_bf16(a0, load_frag_b(kp + (size_t)(j + 16) * HDIM, HDIM), s1);
    s1 = wmma_bf16(a1, load_frag_b(kp + (size_t)(j + 16) * HDIM + 32, HDIM), s1);

    // issue V-fragment loads early: the whole softmax below hides their latency
    v16bf vb[4];
#pragma unroll
    for (int t = 0; t < 4; ++t)
      vb[t] = load_frag_b(vp + (size_t)t * 16 * SEQ + j, SEQ);

    // prefetch next iteration's K tile toward near caches (global_prefetch_b8)
    int jn = (j + 32) & (SEQ - 1);
    __builtin_prefetch(frag_b_ptr(kp + (size_t)jn * HDIM, HDIM), 0, 3);
    __builtin_prefetch(frag_b_ptr(kp + (size_t)(jn + 16) * HDIM, HDIM), 0, 3);

    float m0 = mp[j + n_l], m1 = mp[j + 16 + n_l];
    float tm[8];
#pragma unroll
    for (int r = 0; r < 8; ++r) {
      s0[r] = s0[r] * 0.125f + m0;         // 1/sqrt(64)
      s1[r] = s1[r] * 0.125f + m1;
      tm[r] = fmaxf(s0[r], s1[r]);
    }
#pragma unroll
    for (int r = 0; r < 8; ++r)
#pragma unroll
      for (int mm = 8; mm >= 1; mm >>= 1)
        tm[r] = fmaxf(tm[r], __shfl_xor(tm[r], mm, 32));
    float corr[8];
#pragma unroll
    for (int r = 0; r < 8; ++r) {
      float nm = fmaxf(rmax[r], tm[r]);
      corr[r] = __expf(rmax[r] - nm);
      rmax[r] = nm;
    }
    float ts[8];
#pragma unroll
    for (int r = 0; r < 8; ++r) {
      float p0 = __expf(s0[r] - rmax[r]);
      float p1 = __expf(s1[r] - rmax[r]);
      P[(hi * 8 + r) * 32 + n_l]      = to_bf16(p0);   // C-layout -> LDS row-major
      P[(hi * 8 + r) * 32 + 16 + n_l] = to_bf16(p1);
      ts[r] = p0 + p1;
    }
#pragma unroll
    for (int r = 0; r < 8; ++r) {
#pragma unroll
      for (int mm = 8; mm >= 1; mm >>= 1)
        ts[r] += __shfl_xor(ts[r], mm, 32);
      rsum[r] = rsum[r] * corr[r] + ts[r];
    }
#pragma unroll
    for (int t = 0; t < 4; ++t)
#pragma unroll
      for (int r = 0; r < 8; ++r) o[t][r] *= corr[r];
    // wave-private LDS transpose round-trip: enforce DS completion (CDNA5 split counter)
    asm volatile("s_wait_dscnt 0" ::: "memory");
    v16bf pa = load_frag_a(P, 32);         // A-layout read of P (16x32)
#pragma unroll
    for (int t = 0; t < 4; ++t)
      o[t] = wmma_bf16(pa, vb[t], o[t]);
  }
  int h_ = bh & 15;
#pragma unroll
  for (int r = 0; r < 8; ++r) {
    float inv = 1.f / rsum[r];
    int s_ = qbase + hi * 8 + r;
    size_t base = ((size_t)b_ * SEQ + s_) * HID + h_ * HDIM;
#pragma unroll
    for (int t = 0; t < 4; ++t)
      out[base + t * 16 + n_l] = to_bf16(o[t][r] * inv);
  }
}

// ---------- kernel 5: out-proj GEMM + bias + residual (ping-pong pipeline) ----------

__global__ void __launch_bounds__(256, 1) out_gemm(
    const __bf16* __restrict__ attn, const __bf16* __restrict__ wt,
    const float* __restrict__ bias, const float* __restrict__ resid,
    float* __restrict__ out) {
  int wave = (blockIdx.x * blockDim.x + threadIdx.x) >> 5;
  int mt = wave >> 4;                 // 512 M-tiles
  int ns = wave & 15;                 // 16 N-strips of 64
  int lane = threadIdx.x & 31;
  int n_l = lane & 15, hi = lane >> 4;
  v8f acc[4] = {};
  const __bf16* arow = attn + (size_t)mt * 16 * HID;
  const __bf16* brow = wt + (size_t)ns * 64 * HID;

  Frags f0 = load_step(arow, brow, 0);
#pragma unroll 1
  for (int k0 = 32; k0 < HID - 32; k0 += 64) {
    Frags f1 = load_step(arow, brow, k0);
    step_wmma(acc, f0);
    f0 = load_step(arow, brow, k0 + 32);
    step_wmma(acc, f1);
  }
  {
    Frags f1 = load_step(arow, brow, HID - 32);
    step_wmma(acc, f0);
    step_wmma(acc, f1);
  }

#pragma unroll
  for (int t = 0; t < 4; ++t) {
    int n = ns * 64 + t * 16 + n_l;
    float bz = bias[n];
#pragma unroll
    for (int r = 0; r < 8; ++r) {
      int row = mt * 16 + hi * 8 + r;
      size_t idx = (size_t)row * HID + n;
      out[idx] = acc[t][r] + bz + resid[idx];
    }
  }
}

// ---------- launcher ----------

extern "C" void kernel_launch(void* const* d_in, const int* in_sizes, int n_in,
                              void* d_out, int out_size, void* d_ws, size_t ws_size,
                              hipStream_t stream) {
  const float* hs   = (const float*)d_in[0];
  const float* mask = (const float*)d_in[1];
  const float* g    = (const float*)d_in[2];
  const float* be   = (const float*)d_in[3];
  const float* wqkv = (const float*)d_in[4];
  const float* bqkv = (const float*)d_in[5];
  const float* wout = (const float*)d_in[6];
  const float* bout = (const float*)d_in[7];
  float* out = (float*)d_out;

  const size_t ACT = (size_t)BS * HID;           // 8.39M bf16 elems
  __bf16* ws     = (__bf16*)d_ws;
  __bf16* xln    = ws;
  __bf16* qws    = xln   + ACT;
  __bf16* kws    = qws   + ACT;
  __bf16* vtws   = kws   + ACT;
  __bf16* attnws = vtws  + ACT;
  __bf16* wqkvt  = attnws + ACT;                 // 3.15M
  __bf16* woutt  = wqkvt + (size_t)HID * 3 * HID;// 1.05M  (total ~92 MB)

  cvt_transpose<<<(HID * 3 * HID + 255) / 256, 256, 0, stream>>>(wqkv, wqkvt, HID, 3 * HID);
  cvt_transpose<<<(HID * HID + 255) / 256, 256, 0, stream>>>(wout, woutt, HID, HID);
  ln_kernel<<<BS / 8, 256, 0, stream>>>(hs, g, be, xln);
  qkv_gemm<<<(512 * 48) / 8, 256, 0, stream>>>(xln, wqkvt, bqkv, qws, kws, vtws);
  attn_kernel<<<BATCH * NHEADS * (SEQ / 64), 128, 0, stream>>>(qws, kws, vtws, mask, attnws);
  out_gemm<<<(512 * 16) / 8, 256, 0, stream>>>(attnws, woutt, bout, hs, out);
}
